// TransformerNet_80479097193056
// MI455X (gfx1250) — compile-verified, matrix operand order checked by reading, not run on hardware
//
#include <hip/hip_runtime.h>
#include <hip/hip_bf16.h>
#include <math.h>

// ---------------------------------------------------------------------------
// Types for CDNA5 WMMA (wave32): A/B = 16 x bf16 per lane, C/D = 8 x f32.
// ---------------------------------------------------------------------------
typedef __attribute__((ext_vector_type(16))) __bf16          v16bf;
typedef __attribute__((ext_vector_type(8)))  __bf16          v8bf;
typedef __attribute__((ext_vector_type(8)))  float           v8f;

#define D_IN   128
#define HEADS  4
#define D_H    16
#define HID    64

__device__ __forceinline__ unsigned short f32_to_bf16_rne(float f) {
    unsigned u = __float_as_uint(f);
    unsigned r = (u + 0x7FFFu + ((u >> 16) & 1u)) >> 16;   // RNE
    return (unsigned short)r;
}
// monotone map float -> uint so unsigned atomicMax implements float max
__device__ __forceinline__ unsigned f32_ordered(float f) {
    unsigned u = __float_as_uint(f);
    return (u & 0x80000000u) ? ~u : (u | 0x80000000u);
}
__device__ __forceinline__ float ordered_f32(unsigned u) {
    return (u & 0x80000000u) ? __uint_as_float(u & 0x7FFFFFFFu)
                             : __uint_as_float(~u);
}

// ---------------------------------------------------------------------------
// f32 -> bf16 conversion of the layer input activation matrix (row major)
// ---------------------------------------------------------------------------
__global__ __launch_bounds__(256)
void k_cvt_act(const float* __restrict__ in, unsigned short* __restrict__ out, int n) {
    int i = blockIdx.x * blockDim.x + threadIdx.x;
    if (i < n) out[i] = f32_to_bf16_rne(in[i]);
}

// ---------------------------------------------------------------------------
// Convert the 4 weight matrices (K x 64, row major f32) of one layer into
// bf16 *pre-swizzled into the WMMA B-fragment layout*:
//   out[((w*(K/32) + kbi)*4 + c)*512 + lane*16 + j]
// where lane = hi*16 + nl holds column n = c*16 + nl and
//   j<8  -> k = kbi*32 + hi*8 + j
//   j>=8 -> k = kbi*32 + hi*8 + j + 8
// so each lane's 16-element fragment is one contiguous 32-byte chunk.
// ---------------------------------------------------------------------------
__global__ __launch_bounds__(256)
void k_cvt_w_frag(const float* __restrict__ w0, const float* __restrict__ w1,
                  const float* __restrict__ w2, const float* __restrict__ w3,
                  unsigned short* __restrict__ out, int K) {
    int i = blockIdx.x * blockDim.x + threadIdx.x;
    const int perW = K * HID;
    if (i >= 4 * perW) return;
    const int w = i / perW;
    const int s = i - w * perW;
    const int j    = s & 15;
    const int lane = (s >> 4) & 31;
    const int c    = (s >> 9) & 3;
    const int kbi  = s >> 11;
    const int nl = lane & 15, hi = lane >> 4;
    const int n = c * 16 + nl;
    const int k = kbi * 32 + hi * 8 + ((j < 8) ? j : (j + 8));
    const float* src = (w == 0) ? w0 : (w == 1) ? w1 : (w == 2) ? w2 : w3;
    out[i] = f32_to_bf16_rne(src[k * HID + n]);
}

// ---------------------------------------------------------------------------
// Fused Q/K/V/S GEMM: block = 4 waves; wave w computes outW = A @ Ww + bw for
// one 16-row tile (16x64 output). bf16 WMMA 16x16x32, f32 accumulate.
// A fragments: two contiguous 16B vector loads + register shuffle.
// B fragments: one contiguous 32B vector load (pre-swizzled in global).
// ---------------------------------------------------------------------------
__global__ __launch_bounds__(128)
void k_gemm_qkvs(const unsigned short* __restrict__ abf,
                 const unsigned short* __restrict__ wbf,
                 const float* __restrict__ b0, const float* __restrict__ b1,
                 const float* __restrict__ b2, const float* __restrict__ b3,
                 float* __restrict__ o0, float* __restrict__ o1,
                 float* __restrict__ o2, float* __restrict__ o3,
                 int N, int K) {
    const int tid  = threadIdx.x;
    const int w    = tid >> 5;          // 0..3 : Q,K,V,S
    const int lane = tid & 31;
    const int nl   = lane & 15;
    const int hi   = lane >> 4;

    const int m0 = blockIdx.x * 16;
    int row = m0 + nl;
    if (row >= N) row = N - 1;          // keep EXEC all-ones for WMMA

    // this wave's pre-swizzled weight blob
    const unsigned short* wp = wbf + (size_t)w * K * HID;

    v8f acc[4] = {};

    const int ksteps = K >> 5;
    for (int kbi = 0; kbi < ksteps; ++kbi) {
        // ---- A fragment ----
        const unsigned short* ap = abf + (size_t)row * K + kbi * 32;
        v8bf a0 = *(const v8bf*)(ap + hi * 8);
        v8bf a1 = *(const v8bf*)(ap + 16 + hi * 8);
        v16bf a = __builtin_shufflevector(a0, a1,
                     0, 1, 2, 3, 4, 5, 6, 7, 8, 9, 10, 11, 12, 13, 14, 15);
#pragma unroll
        for (int c = 0; c < 4; ++c) {
            // ---- B fragment: one 32B contiguous chunk per lane ----
            const unsigned short* bp = wp + ((size_t)(kbi * 4 + c) * 32 + lane) * 16;
            v16bf b = *(const v16bf*)bp;
            acc[c] = __builtin_amdgcn_wmma_f32_16x16x32_bf16(
                false, a, false, b, (short)0, acc[c], false, false);
        }
    }

    float*       outp = (w == 0) ? o0 : (w == 1) ? o1 : (w == 2) ? o2 : o3;
    const float* bp   = (w == 0) ? b0 : (w == 1) ? b1 : (w == 2) ? b2 : b3;
#pragma unroll
    for (int c = 0; c < 4; ++c) {
        const float bv = bp[c * 16 + nl];
#pragma unroll
        for (int r = 0; r < 8; ++r) {
            const int rr = m0 + hi * 8 + r;
            if (rr < N) outp[(size_t)rr * HID + c * 16 + nl] = acc[c][r] + bv;
        }
    }
}

// ---------------------------------------------------------------------------
// Edge-attention phase (all f32, L2-resident gathers + atomics)
// ---------------------------------------------------------------------------
__global__ __launch_bounds__(256)
void k_init_attn(unsigned* __restrict__ mxu, float* __restrict__ denom, int n) {
    int i = blockIdx.x * blockDim.x + threadIdx.x;
    if (i < n) { mxu[i] = 0u; denom[i] = 0.0f; }
}

// score[e,h] = 0.25 * <q[dst,h,:], k[src,h,:]>  ; segment max via atomicMax
__global__ __launch_bounds__(256)
void k_score(const float* __restrict__ q, const float* __restrict__ k,
             const int* __restrict__ esrc, const int* __restrict__ edst,
             int E, int ET, float* __restrict__ sc, unsigned* __restrict__ mxu) {
    int t = blockIdx.x * blockDim.x + threadIdx.x;
    if (t >= ET * HEADS) return;
    const int e = t >> 2, h = t & 3;
    const int src = (e < E) ? esrc[e] : (e - E);
    const int dst = (e < E) ? edst[e] : (e - E);
    const float4* qp = (const float4*)(q + (size_t)dst * HID + h * D_H);
    const float4* kp = (const float4*)(k + (size_t)src * HID + h * D_H);
    float s = 0.0f;
#pragma unroll
    for (int i = 0; i < 4; ++i) {
        float4 a = qp[i], b = kp[i];
        s += a.x * b.x + a.y * b.y + a.z * b.z + a.w * b.w;
    }
    s *= 0.25f;                          // 1/sqrt(16)
    sc[t] = s;
    atomicMax(&mxu[dst * HEADS + h], f32_ordered(s));
}

// e = exp(score - max) ; denom += e (segment sum)
__global__ __launch_bounds__(256)
void k_expsum(float* __restrict__ sc, const unsigned* __restrict__ mxu,
              float* __restrict__ denom, const int* __restrict__ edst,
              int E, int ET) {
    int t = blockIdx.x * blockDim.x + threadIdx.x;
    if (t >= ET * HEADS) return;
    const int e = t >> 2, h = t & 3;
    const int dst = (e < E) ? edst[e] : (e - E);
    const float m  = ordered_f32(mxu[dst * HEADS + h]);
    const float ev = expf(sc[t] - m);
    sc[t] = ev;
    atomicAdd(&denom[dst * HEADS + h], ev);
}

// s[dst] += (e/denom[dst]) * v[src]   (accumulates into skip buffer)
__global__ __launch_bounds__(256)
void k_message(const float* __restrict__ sc, const float* __restrict__ denom,
               const float* __restrict__ v,
               const int* __restrict__ esrc, const int* __restrict__ edst,
               int E, int ET, float* __restrict__ sbuf) {
    int t = blockIdx.x * blockDim.x + threadIdx.x;
    if (t >= ET * HEADS) return;
    const int e = t >> 2, h = t & 3;
    const int src = (e < E) ? esrc[e] : (e - E);
    const int dst = (e < E) ? edst[e] : (e - E);
    const float alpha = sc[t] / denom[dst * HEADS + h];
    const float4* vp = (const float4*)(v + (size_t)src * HID + h * D_H);
    float* op = sbuf + (size_t)dst * HID + h * D_H;
#pragma unroll
    for (int i = 0; i < 4; ++i) {
        float4 vv = vp[i];
        atomicAdd(op + i * 4 + 0, alpha * vv.x);
        atomicAdd(op + i * 4 + 1, alpha * vv.y);
        atomicAdd(op + i * 4 + 2, alpha * vv.z);
        atomicAdd(op + i * 4 + 3, alpha * vv.w);
    }
}

__global__ __launch_bounds__(256)
void k_relu(const float* __restrict__ in, float* __restrict__ out, int n) {
    int i = blockIdx.x * blockDim.x + threadIdx.x;
    if (i < n) out[i] = fmaxf(in[i], 0.0f);
}

// final linear HID -> 1
__global__ __launch_bounds__(256)
void k_final(const float* __restrict__ h, const float* __restrict__ Wf,
             const float* __restrict__ bf, float* __restrict__ out, int N) {
    int n = blockIdx.x * blockDim.x + threadIdx.x;
    if (n >= N) return;
    const float4* hp = (const float4*)(h + (size_t)n * HID);
    const float4* wp = (const float4*)Wf;
    float acc = bf[0];
#pragma unroll
    for (int i = 0; i < 16; ++i) {
        float4 a = hp[i], b = wp[i];
        acc += a.x * b.x + a.y * b.y + a.z * b.z + a.w * b.w;
    }
    out[n] = acc;
}

// ---------------------------------------------------------------------------
// Driver
// ---------------------------------------------------------------------------
extern "C" void kernel_launch(void* const* d_in, const int* in_sizes, int n_in,
                              void* d_out, int out_size, void* d_ws, size_t ws_size,
                              hipStream_t stream) {
    const float* x  = (const float*)d_in[0];
    const int*   ei = (const int*)d_in[1];      // int32 (JAX x64 disabled)
    // d_in[2] = edge_weight (unused by TransformerConv)

    const int N  = in_sizes[0] / D_IN;
    const int E  = in_sizes[1] / 2;
    const int ET = E + N;                        // with self loops

    const int* esrc = ei;
    const int* edst = ei + E;

    // params: 3 + layer*8 : Wq,bq,Wk,bk,Wv,bv,Ws,bs ; then Wf, bf
    const float* Wf = (const float*)d_in[3 + 3 * 8 + 0];
    const float* bF = (const float*)d_in[3 + 3 * 8 + 1];

    // ---- carve workspace -------------------------------------------------
    char* p = (char*)d_ws;
    auto carve = [&](size_t bytes) -> void* {
        void* r = (void*)p;
        p += (bytes + 255) & ~(size_t)255;
        return r;
    };
    unsigned short* abf = (unsigned short*)carve((size_t)N * D_IN * 2);
    unsigned short* wbf = (unsigned short*)carve((size_t)4 * D_IN * HID * 2);
    float* qb   = (float*)carve((size_t)N * HID * 4);
    float* kb   = (float*)carve((size_t)N * HID * 4);
    float* vb   = (float*)carve((size_t)N * HID * 4);
    float* sb   = (float*)carve((size_t)N * HID * 4);
    float* sc   = (float*)carve((size_t)ET * HEADS * 4);
    unsigned* mxu  = (unsigned*)carve((size_t)N * HEADS * 4);
    float*    den  = (float*)carve((size_t)N * HEADS * 4);
    float*    hbuf = (float*)carve((size_t)N * HID * 4);

    const int TB = 256;
    const int rowTiles = (N + 15) / 16;
    const int edgeThreads = ET * HEADS;

    for (int layer = 0; layer < 3; ++layer) {
        const int K = (layer == 0) ? D_IN : HID;
        const float* hin = (layer == 0) ? x : hbuf;
        const int base = 3 + layer * 8;
        const float* Wq = (const float*)d_in[base + 0];
        const float* bq = (const float*)d_in[base + 1];
        const float* Wk = (const float*)d_in[base + 2];
        const float* bk2= (const float*)d_in[base + 3];
        const float* Wv = (const float*)d_in[base + 4];
        const float* bv = (const float*)d_in[base + 5];
        const float* Ws = (const float*)d_in[base + 6];
        const float* bs = (const float*)d_in[base + 7];

        // 1) bf16 conversions (weights pre-swizzled to B-fragment layout)
        k_cvt_act<<<((size_t)N * K + TB - 1) / TB, TB, 0, stream>>>(hin, abf, N * K);
        k_cvt_w_frag<<<(4 * K * HID + TB - 1) / TB, TB, 0, stream>>>(Wq, Wk, Wv, Ws, wbf, K);

        // 2) fused Q/K/V/S GEMM (WMMA bf16), bias fused; S is the skip buffer
        k_gemm_qkvs<<<rowTiles, 128, 0, stream>>>(
            abf, wbf, bq, bk2, bv, bs, qb, kb, vb, sb, N, K);

        // 3) attention softmax over incoming edges + scatter into skip buffer
        k_init_attn<<<((size_t)N * HEADS + TB - 1) / TB, TB, 0, stream>>>(mxu, den, N * HEADS);
        k_score<<<((size_t)edgeThreads + TB - 1) / TB, TB, 0, stream>>>(
            qb, kb, esrc, edst, E, ET, sc, mxu);
        k_expsum<<<((size_t)edgeThreads + TB - 1) / TB, TB, 0, stream>>>(
            sc, mxu, den, edst, E, ET);
        k_message<<<((size_t)edgeThreads + TB - 1) / TB, TB, 0, stream>>>(
            sc, den, vb, esrc, edst, E, ET, sb);

        // 4) h = relu(agg + skip)
        k_relu<<<((size_t)N * HID + TB - 1) / TB, TB, 0, stream>>>(sb, hbuf, N * HID);
    }

    // final linear HID -> 1
    k_final<<<(N + TB - 1) / TB, TB, 0, stream>>>(hbuf, Wf, bF, (float*)d_out, N);
}
